// SparsetralGateAdapter_74431783240388
// MI455X (gfx1250) — compile-verified, matrix operand order checked by reading, not run on hardware
//
#include <hip/hip_runtime.h>
#include <hip/hip_bf16.h>
#include <math.h>

typedef __bf16 bf16_t;
typedef bf16_t v16bf __attribute__((ext_vector_type(16)));
typedef bf16_t v8bf  __attribute__((ext_vector_type(8)));
typedef bf16_t v4bf  __attribute__((ext_vector_type(4)));
typedef float  v8f   __attribute__((ext_vector_type(8)));
typedef float  v4f   __attribute__((ext_vector_type(4)));

#define TOKENS 4096
#define HIDDEN 4096
#define ADIM   512
#define NEXP   8
#define BM     128
#define BN     128
#define BK     32
#define LDSK   (BK + 8)   // padded LDS row stride (elements)

// ---------------------------------------------------------------------------
// helpers
// ---------------------------------------------------------------------------

// Branch-free tanh-form GELU: 1 v_exp + 1 v_rcp + a few FMAs, saturates at +/-1.
__device__ __forceinline__ float gelu_fast(float x) {
    const float u = 0.7978845608028654f * fmaf(0.044715f * x * x, x, x);
    const float e = __expf(2.0f * u);
    const float t = 1.0f - 2.0f * __builtin_amdgcn_rcpf(e + 1.0f); // tanh(u)
    return 0.5f * x * (1.0f + t);
}

// 16-byte global -> LDS asynchronous copy (ASYNCcnt-tracked, no VGPR bounce).
// LDS byte address = low 32 bits of the flat pointer.
__device__ __forceinline__ void async_copy16(void* lds_ptr, const void* gptr) {
    const unsigned int       ldsa = (unsigned int)(uintptr_t)lds_ptr;
    const unsigned long long ga   = (unsigned long long)(uintptr_t)gptr;
    asm volatile("global_load_async_to_lds_b128 %0, %1, off"
                 :: "v"(ldsa), "v"(ga) : "memory");
}
__device__ __forceinline__ void wait_async_le4() {
    asm volatile("s_wait_asynccnt 0x4" ::: "memory");
}
__device__ __forceinline__ void wait_async_0() {
    asm volatile("s_wait_asynccnt 0x0" ::: "memory");
}

// 16x32 bf16 WMMA operand fragment: lane = (m%16, k = 8*(lane/16) + {0..7, 16..23}),
// two contiguous 16B chunks of a row-major 32-element LDS row.
__device__ __forceinline__ v16bf load_frag(const bf16_t* base) {
    v8bf lo = *reinterpret_cast<const v8bf*>(base);
    v8bf hi = *reinterpret_cast<const v8bf*>(base + 16);
    return __builtin_shufflevector(lo, hi, 0,1,2,3,4,5,6,7,8,9,10,11,12,13,14,15);
}

__device__ __forceinline__ v8f wmma_bf16(v16bf a, v16bf b, v8f c) {
    return __builtin_amdgcn_wmma_f32_16x16x32_bf16(
        false, a, false, b, (short)0, c, false, false);
}

// ---------------------------------------------------------------------------
// Pre-pass 1: f32 -> bf16 straight cast (for X)
// ---------------------------------------------------------------------------
__global__ __launch_bounds__(256) void cast_bf16_kernel(
    const float* __restrict__ src, bf16_t* __restrict__ dst)
{
    const size_t i = ((size_t)blockIdx.x * 256 + threadIdx.x) * 4;
    const v4f v = *reinterpret_cast<const v4f*>(src + i);
    v4bf b; b.x = (bf16_t)v.x; b.y = (bf16_t)v.y; b.z = (bf16_t)v.z; b.w = (bf16_t)v.w;
    *reinterpret_cast<v4bf*>(dst + i) = b;
}

// ---------------------------------------------------------------------------
// Pre-pass 2: transpose + cast weights to bf16 [N][K] (K = 4096 both views).
// mode=1: src = w_down [E,D,A], flat B(k=d, n=e*512+a)  (ld = 512 within expert)
// mode=0: src = w_up   [E,A,D] = flat [K=4096, N=4096]  (ld = 4096)
// ---------------------------------------------------------------------------
__global__ __launch_bounds__(256) void transpose_cast_kernel(
    const float* __restrict__ src, bf16_t* __restrict__ dst, int mode)
{
    __shared__ float tile[32][33];
    const int k0 = blockIdx.x * 32;
    const int n0 = blockIdx.y * 32;
    const float* s; int ld;
    if (mode) { s = src + (size_t)(n0 >> 9) * HIDDEN * ADIM + (n0 & (ADIM - 1)); ld = ADIM; }
    else      { s = src + n0;                                                    ld = HIDDEN; }
    const int tx = threadIdx.x & 31, ty = threadIdx.x >> 5;
#pragma unroll
    for (int i = 0; i < 4; ++i) {
        const int k = ty + i * 8;
        tile[k][tx] = s[(size_t)(k0 + k) * ld + tx];
    }
    __syncthreads();
    const int row = threadIdx.x >> 3, g = threadIdx.x & 7;
    v4bf b;
    b.x = (bf16_t)tile[g * 4 + 0][row];
    b.y = (bf16_t)tile[g * 4 + 1][row];
    b.z = (bf16_t)tile[g * 4 + 2][row];
    b.w = (bf16_t)tile[g * 4 + 3][row];
    *reinterpret_cast<v4bf*>(dst + (size_t)(n0 + row) * HIDDEN + k0 + g * 4) = b;
}

// ---------------------------------------------------------------------------
// Router: one wave per token; fp32 softmax, top-2, renormalize.
// ---------------------------------------------------------------------------
__global__ __launch_bounds__(256) void router_kernel(
    const float* __restrict__ rh, const float* __restrict__ wr,
    float* __restrict__ wfull, float* __restrict__ rowsum)
{
    const int wave = threadIdx.x >> 5;
    const int lane = threadIdx.x & 31;
    const int t = blockIdx.x * 8 + wave;
    const float* x = rh + (size_t)t * HIDDEN;

    float acc[NEXP];
#pragma unroll
    for (int e = 0; e < NEXP; ++e) acc[e] = 0.0f;
    for (int d = lane; d < HIDDEN; d += 32) {
        const float xv = x[d];
        const float* w = wr + (size_t)d * NEXP;
#pragma unroll
        for (int e = 0; e < NEXP; ++e) acc[e] = fmaf(xv, w[e], acc[e]);
    }
#pragma unroll
    for (int e = 0; e < NEXP; ++e) {
#pragma unroll
        for (int off = 16; off > 0; off >>= 1)
            acc[e] += __shfl_xor(acc[e], off, 32);
    }
    if (lane == 0) {
        float m = acc[0];
#pragma unroll
        for (int e = 1; e < NEXP; ++e) m = fmaxf(m, acc[e]);
        float p[NEXP], s = 0.0f;
#pragma unroll
        for (int e = 0; e < NEXP; ++e) { p[e] = __expf(acc[e] - m); s += p[e]; }
        const float inv = 1.0f / s;
#pragma unroll
        for (int e = 0; e < NEXP; ++e) p[e] *= inv;
        int i1 = 0; float p1 = p[0];
#pragma unroll
        for (int e = 1; e < NEXP; ++e) if (p[e] > p1) { p1 = p[e]; i1 = e; }
        int i2 = -1; float p2 = -1.0f;
#pragma unroll
        for (int e = 0; e < NEXP; ++e) if (e != i1 && p[e] > p2) { p2 = p[e]; i2 = e; }
        const float wsum = p1 + p2;
        const float w1 = p1 / wsum, w2 = p2 / wsum;
#pragma unroll
        for (int e = 0; e < NEXP; ++e)
            wfull[(size_t)t * NEXP + e] = (e == i1) ? w1 : ((e == i2) ? w2 : 0.0f);
        rowsum[t] = w1 + w2;
    }
}

// ---------------------------------------------------------------------------
// Unified 4096^3 bf16 WMMA GEMM, double-buffered async-DMA LDS staging.
// A  [4096][4096] bf16 row-major, Bt [N][K] bf16 row-major (pre-transposed).
// MODE 0: Hp[t,n] = wfull[t, n>>9] * gelu(acc)        (bf16 out)
// MODE 1: out[t,n] = acc + OH[t,n] * rowsum[t]        (f32 out)
// ---------------------------------------------------------------------------
template <int MODE>
__global__ __launch_bounds__(256) void gemm_kernel(
    const bf16_t* __restrict__ A, const bf16_t* __restrict__ Bt,
    const float* __restrict__ scale, const float* __restrict__ OH,
    bf16_t* __restrict__ outb, float* __restrict__ outf)
{
    __shared__ alignas(16) bf16_t As[2][BM * LDSK];
    __shared__ alignas(16) bf16_t Bs[2][BN * LDSK];
    __shared__ float sw[BM];

    const int tid = threadIdx.x;
    const int r0 = blockIdx.x * BM;
    const int c0 = blockIdx.y * BN;

    if (tid < BM)
        sw[tid] = (MODE == 0) ? scale[(size_t)(r0 + tid) * NEXP + (c0 >> 9)]
                              : scale[r0 + tid];

    const int wave = tid >> 5, lane = tid & 31;
    const int wm = wave >> 2, wn = wave & 3;   // 2 x 4 wave grid -> 64x32 per wave
    const int lm = lane & 15, lh = lane >> 4;

    // per-thread staging coords: 2 chunks of 16B for A, 2 for B, per tile
    const int ar0 = (tid)       >> 2, aq0 = (tid)       & 3;
    const int ar1 = (tid + 256) >> 2, aq1 = (tid + 256) & 3;

    v8f acc[4][2];
#pragma unroll
    for (int i = 0; i < 4; ++i)
#pragma unroll
        for (int j = 0; j < 2; ++j) acc[i][j] = v8f{};

    // ---- issue tile 0 into buffer 0 ----
    {
        async_copy16(&As[0][ar0 * LDSK + aq0 * 8], A  + (size_t)(r0 + ar0) * HIDDEN + aq0 * 8);
        async_copy16(&As[0][ar1 * LDSK + aq1 * 8], A  + (size_t)(r0 + ar1) * HIDDEN + aq1 * 8);
        async_copy16(&Bs[0][ar0 * LDSK + aq0 * 8], Bt + (size_t)(c0 + ar0) * HIDDEN + aq0 * 8);
        async_copy16(&Bs[0][ar1 * LDSK + aq1 * 8], Bt + (size_t)(c0 + ar1) * HIDDEN + aq1 * 8);
    }

    for (int k0 = 0; k0 < HIDDEN; k0 += BK) {
        const int p = (k0 >> 5) & 1;
        if (k0 + BK < HIDDEN) {
            const int kn = k0 + BK;
            async_copy16(&As[p ^ 1][ar0 * LDSK + aq0 * 8], A  + (size_t)(r0 + ar0) * HIDDEN + kn + aq0 * 8);
            async_copy16(&As[p ^ 1][ar1 * LDSK + aq1 * 8], A  + (size_t)(r0 + ar1) * HIDDEN + kn + aq1 * 8);
            async_copy16(&Bs[p ^ 1][ar0 * LDSK + aq0 * 8], Bt + (size_t)(c0 + ar0) * HIDDEN + kn + aq0 * 8);
            async_copy16(&Bs[p ^ 1][ar1 * LDSK + aq1 * 8], Bt + (size_t)(c0 + ar1) * HIDDEN + kn + aq1 * 8);
            wait_async_le4();       // tile k0 resident; next 4 still in flight
        } else {
            wait_async_0();
        }
        __syncthreads();

        v16bf af[4], bf[2];
#pragma unroll
        for (int i = 0; i < 4; ++i)
            af[i] = load_frag(&As[p][(wm * 64 + i * 16 + lm) * LDSK + lh * 8]);
#pragma unroll
        for (int j = 0; j < 2; ++j)
            bf[j] = load_frag(&Bs[p][(wn * 32 + j * 16 + lm) * LDSK + lh * 8]);
#pragma unroll
        for (int i = 0; i < 4; ++i)
#pragma unroll
            for (int j = 0; j < 2; ++j)
                acc[i][j] = wmma_bf16(af[i], bf[j], acc[i][j]);

        __syncthreads();   // all reads of buffer p done before it is re-targeted
    }

    // ---- epilogue ----
#pragma unroll
    for (int i = 0; i < 4; ++i) {
#pragma unroll
        for (int j = 0; j < 2; ++j) {
            const int gcol = c0 + wn * 32 + j * 16 + lm;
#pragma unroll
            for (int r = 0; r < 8; ++r) {
                const int lrow = wm * 64 + i * 16 + r + 8 * lh;
                if (MODE == 0) {
                    const float g = gelu_fast(acc[i][j][r]) * sw[lrow];
                    outb[(size_t)(r0 + lrow) * HIDDEN + gcol] = (bf16_t)g;
                } else {
                    const size_t idx = (size_t)(r0 + lrow) * HIDDEN + gcol;
                    outf[idx] = acc[i][j][r] + OH[idx] * sw[lrow];
                }
            }
        }
    }
}

// ---------------------------------------------------------------------------
// launch
// ---------------------------------------------------------------------------
extern "C" void kernel_launch(void* const* d_in, const int* in_sizes, int n_in,
                              void* d_out, int out_size, void* d_ws, size_t ws_size,
                              hipStream_t stream) {
    (void)in_sizes; (void)n_in; (void)out_size; (void)ws_size;
    const float* X  = (const float*)d_in[0];
    const float* OH = (const float*)d_in[1];
    const float* RH = (const float*)d_in[2];
    const float* WD = (const float*)d_in[3];
    const float* WU = (const float*)d_in[4];
    const float* WR = (const float*)d_in[5];
    float* out = (float*)d_out;

    const size_t MAT = (size_t)4096 * 4096;           // elements per 4096x4096 matrix
    char* ws = (char*)d_ws;
    bf16_t* Hp    = (bf16_t*)ws;                      // [T, E*A] bf16
    bf16_t* Xb    = Hp  + MAT;                        // X as bf16
    bf16_t* WdT   = Xb  + MAT;                        // w_down^T  [n=e*512+a][k=d]
    bf16_t* WuT   = WdT + MAT;                        // w_up^T    [n=d][k=e*512+a]
    float*  wfull = (float*)(WuT + MAT);
    float*  rsum  = wfull + (size_t)TOKENS * NEXP;

    cast_bf16_kernel<<<(int)(MAT / (256 * 4)), 256, 0, stream>>>(X, Xb);
    dim3 gt(4096 / 32, 4096 / 32);
    transpose_cast_kernel<<<gt, 256, 0, stream>>>(WD, WdT, 1);
    transpose_cast_kernel<<<gt, 256, 0, stream>>>(WU, WuT, 0);
    router_kernel<<<TOKENS / 8, 256, 0, stream>>>(RH, WR, wfull, rsum);

    dim3 gg(TOKENS / BM, 4096 / BN);
    gemm_kernel<0><<<gg, 256, 0, stream>>>(Xb, WdT, wfull, nullptr, Hp, nullptr);
    gemm_kernel<1><<<gg, 256, 0, stream>>>(Hp, WuT, rsum, OH, nullptr, out);
}
